// ConvTransformerEncoder_17849884082850
// MI455X (gfx1250) — compile-verified, matrix operand order checked by reading, not run on hardware
//
#include <hip/hip_runtime.h>
#include <hip/hip_bf16.h>
#include <math.h>

// ---------------------------------------------------------------------------
// ConvTransformerEncoder forward for MI455X (gfx1250, wave32, WMMA).
// All GEMMs: v_wmma_f32_16x16x32_f16, one wave = one 16x16 C tile.
// Weights pre-transposed to f16 (N,K) so every A/B tile load is 2x b128.
// 16x16 tile stores staged through per-wave LDS for coalesced b128 stores.
// ---------------------------------------------------------------------------

typedef __attribute__((ext_vector_type(16))) _Float16 v16h;
typedef __attribute__((ext_vector_type(8)))  _Float16 v8h;
typedef __attribute__((ext_vector_type(4)))  _Float16 v4h;
typedef __attribute__((ext_vector_type(8)))  float    v8f;

#define B_    8
#define WIN_  1024
#define DIN_  8
#define DM_   128
#define NH_   8
#define DH_   1024     // NH_ * DM_
#define DHID_ 512
#define NL_   2
#define ROWS_ (B_ * WIN_)   // 8192

// ---------------- WMMA helpers (wave32 VGPR layouts per CDNA5 ISA 7.12.2) ---

__device__ __forceinline__ v8f wmma_f16(v16h a, v16h b, v8f c) {
  return __builtin_amdgcn_wmma_f32_16x16x32_f16(
      /*neg_a=*/false, a, /*neg_b=*/false, b,
      /*c_mod=*/(short)0, c, /*reuse_a=*/false, /*reuse_b=*/false);
}

// 16x32 (MxK) A tile from row-major f16, leading dim ld. Also used for B when
// B is stored transposed (N,K) row-major: identical lane layout with N<->M.
// lane<16: rows M=lane, K = 0..7 & 16..23 ; lane>=16: same rows, K = 8..15 & 24..31
__device__ __forceinline__ v16h load_a_f16(const _Float16* p, int ld) {
  int lane = threadIdx.x & 31;
  int m  = lane & 15;
  int kb = (lane >> 4) * 8;
  const _Float16* row = p + (size_t)m * ld + kb;
  v16h a;
#pragma unroll
  for (int j = 0; j < 8; ++j) a[j]     = row[j];        // 16B
#pragma unroll
  for (int j = 0; j < 8; ++j) a[8 + j] = row[16 + j];   // 16B
  return a;
}

// A tile from f32 source (softmax weights), converted on the fly.
__device__ __forceinline__ v16h load_a_f32(const float* p, int ld) {
  int lane = threadIdx.x & 31;
  int m  = lane & 15;
  int kb = (lane >> 4) * 8;
  const float* row = p + (size_t)m * ld + kb;
  v16h a;
#pragma unroll
  for (int j = 0; j < 8; ++j) a[j]     = (_Float16)row[j];
#pragma unroll
  for (int j = 0; j < 8; ++j) a[8 + j] = (_Float16)row[16 + j];
  return a;
}

// C/D layout: VGPR i -> M = i + 8*(lane>=16), N = lane&15.

// Stage a 16x16 f16 tile through wave-private LDS, emit coalesced 16B stores.
// lds256: 256 halfs private to this wave. dst: tile origin, leading dim ld.
__device__ __forceinline__ void store_tile_f16_lds(_Float16* lds256, _Float16* dst,
                                                   int ld, const v8f& vals) {
  int lane = threadIdx.x & 31;
  int n = lane & 15, mb = (lane >> 4) * 8;
#pragma unroll
  for (int i = 0; i < 8; ++i) lds256[(mb + i) * 16 + n] = (_Float16)vals[i];
  __builtin_amdgcn_wave_barrier();     // LDS pipe is in-order within a wave
  int r = lane & 15, half = (lane >> 4) * 8;
  v8h row = *(const v8h*)(lds256 + r * 16 + half);
  *(v8h*)(dst + (size_t)r * ld + half) = row;
}

// ---------------- mask (log-sparse attention pattern) -----------------------
// sub_len=64 -> log_l=6 ; rows < 192 are plain causal.
__global__ void mask_kernel(unsigned char* mask) {
  int idx = blockIdx.x * blockDim.x + threadIdx.x;   // 1024*1024 threads
  int r = idx >> 10, c = idx & 1023;
  bool m;
  if (r < 192) {
    m = (c <= r);
  } else {
    m = false;
#pragma unroll
    for (int dj = 0; dj < 6; ++dj) {            // interval [j-5, j]
      int j = c + dj;
      if (j >= 5 && j <= r && ((r - j) & 63) == 0) m = true;
    }
#pragma unroll
    for (int i = 0; i < 6; ++i) {               // log points j-5-2^i
      int j = c + 5 + (1 << i);
      if (j >= 5 && j <= r && ((r - j) & 63) == 0) m = true;
    }
    int j0 = r & 63;                            // final short iteration
    if (j0 <= 4 && c < j0) m = true;
  }
  mask[idx] = m ? 1 : 0;
}

// ---------------- weight prep: (K,N) f32 -> (N,K) f16, LDS-tiled ------------
__global__ void transpose_cvt_kernel(const float* __restrict__ in,
                                     _Float16* __restrict__ outp, int K, int N) {
  __shared__ float tile[32][33];
  int k0 = blockIdx.y << 5, n0 = blockIdx.x << 5;
  int tx = threadIdx.x & 31, ty = threadIdx.x >> 5;   // 256 threads: ty 0..7
#pragma unroll
  for (int i = ty; i < 32; i += 8)
    tile[i][tx] = in[(size_t)(k0 + i) * N + (n0 + tx)];
  __syncthreads();
#pragma unroll
  for (int i = ty; i < 32; i += 8)
    outp[(size_t)(n0 + i) * K + (k0 + tx)] = (_Float16)tile[tx][i];
}

// ---------------- embedding: x = enc @ Wp + bp + PE -------------------------
__global__ void embed_kernel(const float* __restrict__ enc, const float* __restrict__ Wp,
                             const float* __restrict__ bp, float* __restrict__ x,
                             _Float16* __restrict__ xh) {
  int idx = blockIdx.x * blockDim.x + threadIdx.x;   // ROWS_*DM_
  int r = idx >> 7, d = idx & 127;
  int t = r & (WIN_ - 1);
  float acc = bp[d];
  const float* e = enc + (size_t)r * DIN_;
#pragma unroll
  for (int i = 0; i < DIN_; ++i) acc = fmaf(e[i], Wp[i * DM_ + d], acc);
  float div = __expf((float)(d & ~1) * (-9.210340371976184f / (float)DM_));
  acc += (d & 1) ? __cosf((float)t * div) : __sinf((float)t * div);
  x[idx]  = acc;
  xh[idx] = (_Float16)acc;
}

// ---------------- V projection -> transposed v_t[b,h,d,t] -------------------
__global__ void wmma_vproj(const _Float16* __restrict__ xh, const _Float16* __restrict__ Wvt,
                           const float* __restrict__ bv, _Float16* __restrict__ vt) {
  const int NT = DH_ / 16;                        // 64
  int wave = blockIdx.x * (blockDim.x >> 5) + (threadIdx.x >> 5);
  if (wave >= (ROWS_ / 16) * NT) return;
  int tm = wave / NT, tn = wave % NT;
  int m0 = tm << 4, n0 = tn << 4;
  v8f c = {};
#pragma unroll
  for (int kk = 0; kk < DM_; kk += 32)
    c = wmma_f16(load_a_f16(xh + (size_t)m0 * DM_ + kk, DM_),
                 load_a_f16(Wvt + (size_t)n0 * DM_ + kk, DM_), c);
  int b  = m0 >> 10, t0 = m0 & (WIN_ - 1);
  int h  = n0 >> 7,  d0 = n0 & 127;
  int lane = threadIdx.x & 31;
  int n = lane & 15, mb = (lane >> 4) * 8;
  float bias = bv[n0 + n];
  // transposed store: lane n owns row d = d0+n, 8 consecutive t -> one 16B store
  v8h o;
#pragma unroll
  for (int i = 0; i < 8; ++i) o[i] = (_Float16)(c[i] + bias);
  *(v8h*)(vt + ((size_t)(b * NH_ + h) * DM_ + d0 + n) * WIN_ + t0 + mb) = o;
}

// ---------------- QK causal-conv projection (q_len = 3) ---------------------
__device__ __forceinline__ v16h load_a_qk(const _Float16* xb, int t0, int kk, int w) {
  int lane = threadIdx.x & 31;
  int m  = lane & 15;
  int kb = (lane >> 4) * 8;
  int t  = t0 + m + w - 2;                        // left pad by q_len-1
  v16h a = {};
  if (t >= 0) {
    const _Float16* row = xb + (size_t)t * DM_ + kk + kb;
#pragma unroll
    for (int j = 0; j < 8; ++j) a[j]     = row[j];
#pragma unroll
    for (int j = 0; j < 8; ++j) a[8 + j] = row[16 + j];
  }
  return a;
}

__global__ void wmma_qkproj(const _Float16* __restrict__ xh, const _Float16* __restrict__ Wqkt,
                            const float* __restrict__ bqk, _Float16* __restrict__ qbuf,
                            _Float16* __restrict__ kbuf) {
  __shared__ __align__(16) _Float16 stile[8][256];
  const int NT = (2 * DH_) / 16;                  // 128
  int wave = blockIdx.x * (blockDim.x >> 5) + (threadIdx.x >> 5);
  if (wave >= (ROWS_ / 16) * NT) return;
  int tm = wave / NT, tn = wave % NT;
  int m0 = tm << 4, n0 = tn << 4;
  int b  = m0 >> 10, t0 = m0 & (WIN_ - 1);
  const _Float16* xb = xh + (size_t)b * WIN_ * DM_;
  v8f c = {};
#pragma unroll
  for (int w = 0; w < 3; ++w) {
    const _Float16* Bt = Wqkt + (size_t)w * 2 * DH_ * DM_;   // tap w: (2048,128) f16
#pragma unroll
    for (int kk = 0; kk < DM_; kk += 32)
      c = wmma_f16(load_a_qk(xb, t0, kk, w),
                   load_a_f16(Bt + (size_t)n0 * DM_ + kk, DM_), c);
  }
  int nn = n0 & (DH_ - 1);
  int h  = nn >> 7, d0 = nn & 127;
  _Float16* dst = (n0 < DH_ ? qbuf : kbuf)
                + (size_t)(b * NH_ + h) * WIN_ * DM_ + (size_t)t0 * DM_ + d0;
  int lane = threadIdx.x & 31;
  float bias = bqk[n0 + (lane & 15)];
  v8f vals;
#pragma unroll
  for (int i = 0; i < 8; ++i) vals[i] = c[i] + bias;
  store_tile_f16_lds(stile[threadIdx.x >> 5], dst, DM_, vals);
}

// ---------------- scores = scale * q k^T, masked, -> d_out attn region ------
__global__ void wmma_scores(const _Float16* __restrict__ q, const _Float16* __restrict__ k,
                            const unsigned char* __restrict__ mask,
                            float* __restrict__ attn, int l) {
  __shared__ __align__(16) float ftile[8][256];
  const int TPB = (WIN_ / 16) * (WIN_ / 16);      // 4096 tiles per (b,h)
  int wave = blockIdx.x * (blockDim.x >> 5) + (threadIdx.x >> 5);
  if (wave >= B_ * NH_ * TPB) return;
  int bh = wave / TPB, t = wave % TPB;
  int tm = t >> 6, tn = t & 63;
  int m0 = tm << 4, n0 = tn << 4;
  const _Float16* qbh = q + (size_t)bh * WIN_ * DM_;
  const _Float16* kbh = k + (size_t)bh * WIN_ * DM_;
  v8f c = {};
#pragma unroll
  for (int kk = 0; kk < DM_; kk += 32)
    // kbh is (N=1024, K=128) row-major == B^T: same loader as A.
    c = wmma_f16(load_a_f16(qbh + (size_t)m0 * DM_ + kk, DM_),
                 load_a_f16(kbh + (size_t)n0 * DM_ + kk, DM_), c);
  int b = bh >> 3, h = bh & 7;
  float* out = attn + ((size_t)((b * NL_ + l) * NH_ + h) << 20)
             + ((size_t)m0 << 10) + n0;
  const float scale = 0.08838834764831845f;       // 1/sqrt(128)
  int lane = threadIdx.x & 31;
  int n = lane & 15, mb = (lane >> 4) * 8;
  float* lds = ftile[threadIdx.x >> 5];
#pragma unroll
  for (int i = 0; i < 8; ++i) lds[(mb + i) * 16 + n] = c[i] * scale;
  __builtin_amdgcn_wave_barrier();
  int r = lane & 15, half = (lane >> 4) * 8;
  const unsigned char* mrow = mask + (((size_t)(m0 + r)) << 10) + n0 + half;
#pragma unroll
  for (int rep = 0; rep < 2; ++rep) {
    float4 v4 = *(const float4*)(lds + r * 16 + half + rep * 4);
    uchar4 mk = *(const uchar4*)(mrow + rep * 4);
    v4.x = mk.x ? v4.x : -1e9f;
    v4.y = mk.y ? v4.y : -1e9f;
    v4.z = mk.z ? v4.z : -1e9f;
    v4.w = mk.w ? v4.w : -1e9f;
    *(float4*)(out + (size_t)r * WIN_ + half + rep * 4) = v4;
  }
}

// ---------------- row softmax over 1024 (one wave32 per row, float4) --------
__global__ void softmax_kernel(float* __restrict__ attn, int l) {
  int wave = blockIdx.x * (blockDim.x >> 5) + (threadIdx.x >> 5);
  if (wave >= B_ * NH_ * WIN_) return;
  int lane = threadIdx.x & 31;
  int bh = wave >> 10, row = wave & (WIN_ - 1);
  int b = bh >> 3, h = bh & 7;
  float4* p = (float4*)(attn + ((size_t)((b * NL_ + l) * NH_ + h) << 20)
                             + ((size_t)row << 10));
  float4 vv[8];
  float mx = -3.0e38f;
#pragma unroll
  for (int i = 0; i < 8; ++i) {
    vv[i] = p[lane + (i << 5)];
    mx = fmaxf(mx, fmaxf(fmaxf(vv[i].x, vv[i].y), fmaxf(vv[i].z, vv[i].w)));
  }
#pragma unroll
  for (int o = 16; o > 0; o >>= 1) mx = fmaxf(mx, __shfl_xor(mx, o, 32));
  float sum = 0.f;
#pragma unroll
  for (int i = 0; i < 8; ++i) {
    vv[i].x = __expf(vv[i].x - mx); vv[i].y = __expf(vv[i].y - mx);
    vv[i].z = __expf(vv[i].z - mx); vv[i].w = __expf(vv[i].w - mx);
    sum += vv[i].x + vv[i].y + vv[i].z + vv[i].w;
  }
#pragma unroll
  for (int o = 16; o > 0; o >>= 1) sum += __shfl_xor(sum, o, 32);
  float inv = 1.0f / sum;
#pragma unroll
  for (int i = 0; i < 8; ++i) {
    vv[i].x *= inv; vv[i].y *= inv; vv[i].z *= inv; vv[i].w *= inv;
    p[lane + (i << 5)] = vv[i];
  }
}

// ---------------- a = w @ v_t  (A: f32 cvt-on-load; B: transposed v) --------
__global__ void wmma_av(const float* __restrict__ attn, const _Float16* __restrict__ vt,
                        _Float16* __restrict__ abuf, int l) {
  __shared__ __align__(16) _Float16 stile[8][256];
  const int TPB = (WIN_ / 16) * (DM_ / 16);       // 512 tiles per (b,h)
  int wave = blockIdx.x * (blockDim.x >> 5) + (threadIdx.x >> 5);
  if (wave >= B_ * NH_ * TPB) return;
  int bh = wave / TPB, t = wave % TPB;
  int tm = t >> 3, tn = t & 7;
  int m0 = tm << 4, n0 = tn << 4;
  int b = bh >> 3, h = bh & 7;
  const float*    w   = attn + ((size_t)((b * NL_ + l) * NH_ + h) << 20);
  const _Float16* vbh = vt + (size_t)bh * DM_ * WIN_;   // (d, t) row-major
  v8f c = {};
#pragma unroll 4
  for (int kk = 0; kk < WIN_; kk += 32) {
    if (kk + 256 < WIN_)
      __builtin_prefetch(w + ((size_t)m0 << 10) + kk + 256, 0, 1);
    c = wmma_f16(load_a_f32(w + ((size_t)m0 << 10) + kk, WIN_),
                 load_a_f16(vbh + (size_t)n0 * WIN_ + kk, WIN_), c);
  }
  _Float16* dst = abuf + ((size_t)(b * WIN_ + m0)) * DH_ + h * DM_ + n0;
  store_tile_f16_lds(stile[threadIdx.x >> 5], dst, DH_, c);
}

// ---------------- generic WMMA GEMM: C = A(f16) @ Bt(f16 N,K) + bias --------
__global__ void wmma_gemm_f32out(const _Float16* __restrict__ A, const _Float16* __restrict__ Bt,
                                 const float* __restrict__ bias, float* __restrict__ C,
                                 int M, int N, int K) {
  int NT = N >> 4;
  int wave = blockIdx.x * (blockDim.x >> 5) + (threadIdx.x >> 5);
  if (wave >= (M >> 4) * NT) return;
  int tm = wave / NT, tn = wave % NT;
  int m0 = tm << 4, n0 = tn << 4;
  v8f c = {};
  for (int kk = 0; kk < K; kk += 32)
    c = wmma_f16(load_a_f16(A + (size_t)m0 * K + kk, K),
                 load_a_f16(Bt + (size_t)n0 * K + kk, K), c);
  int lane = threadIdx.x & 31;
  int n = lane & 15, mb = (lane >> 4) * 8;
  float bi = bias[n0 + n];
#pragma unroll
  for (int i = 0; i < 8; ++i)
    C[(size_t)(m0 + mb + i) * N + n0 + n] = c[i] + bi;
}

__global__ void wmma_gemm_relu_f16out(const _Float16* __restrict__ A, const _Float16* __restrict__ Bt,
                                      const float* __restrict__ bias, _Float16* __restrict__ C,
                                      int M, int N, int K) {
  __shared__ __align__(16) _Float16 stile[8][256];
  int NT = N >> 4;
  int wave = blockIdx.x * (blockDim.x >> 5) + (threadIdx.x >> 5);
  if (wave >= (M >> 4) * NT) return;
  int tm = wave / NT, tn = wave % NT;
  int m0 = tm << 4, n0 = tn << 4;
  v8f c = {};
  for (int kk = 0; kk < K; kk += 32)
    c = wmma_f16(load_a_f16(A + (size_t)m0 * K + kk, K),
                 load_a_f16(Bt + (size_t)n0 * K + kk, K), c);
  int lane = threadIdx.x & 31;
  float bi = bias[n0 + (lane & 15)];
  v8f vals;
#pragma unroll
  for (int i = 0; i < 8; ++i) vals[i] = fmaxf(c[i] + bi, 0.f);
  store_tile_f16_lds(stile[threadIdx.x >> 5], C + (size_t)m0 * N + n0, N, vals);
}

// ---------------- x = LayerNorm(x + res) * g + b  (one wave per row) --------
__global__ void add_ln_kernel(float* __restrict__ x, const float* __restrict__ res,
                              const float* __restrict__ g, const float* __restrict__ be,
                              _Float16* __restrict__ xh) {
  int wave = blockIdx.x * (blockDim.x >> 5) + (threadIdx.x >> 5);
  if (wave >= ROWS_) return;
  int lane = threadIdx.x & 31;
  float4* xr = (float4*)(x + (size_t)wave * DM_);
  const float4* rr = (const float4*)(res + (size_t)wave * DM_);
  float4 v = xr[lane], rv = rr[lane];
  v.x += rv.x; v.y += rv.y; v.z += rv.z; v.w += rv.w;
  float s = v.x + v.y + v.z + v.w;
#pragma unroll
  for (int o = 16; o > 0; o >>= 1) s += __shfl_xor(s, o, 32);
  float mean = s * (1.f / DM_);
  float dx = v.x - mean, dy = v.y - mean, dz = v.z - mean, dw = v.w - mean;
  float vs = dx * dx + dy * dy + dz * dz + dw * dw;
#pragma unroll
  for (int o = 16; o > 0; o >>= 1) vs += __shfl_xor(vs, o, 32);
  float inv = rsqrtf(vs * (1.f / DM_) + 1e-5f);
  float4 gv = ((const float4*)g)[lane], bv = ((const float4*)be)[lane];
  float4 y;
  y.x = dx * inv * gv.x + bv.x;
  y.y = dy * inv * gv.y + bv.y;
  y.z = dz * inv * gv.z + bv.z;
  y.w = dw * inv * gv.w + bv.w;
  xr[lane] = y;
  v4h hy; hy[0] = (_Float16)y.x; hy[1] = (_Float16)y.y;
  hy[2] = (_Float16)y.z; hy[3] = (_Float16)y.w;
  *(v4h*)(xh + (size_t)wave * DM_ + lane * 4) = hy;
}

// ---------------- out = x @ Wout + bout  (D_OUT = 1) ------------------------
__global__ void out_kernel(const float* __restrict__ x, const float* __restrict__ Wout,
                           const float* __restrict__ bout, float* __restrict__ out) {
  int r = blockIdx.x * blockDim.x + threadIdx.x;
  if (r >= ROWS_) return;
  float acc = bout[0];
  const float* row = x + (size_t)r * DM_;
#pragma unroll 8
  for (int d = 0; d < DM_; ++d) acc = fmaf(row[d], Wout[d], acc);
  out[r] = acc;
}

// ---------------------------------------------------------------------------
extern "C" void kernel_launch(void* const* d_in, const int* in_sizes, int n_in,
                              void* d_out, int out_size, void* d_ws, size_t ws_size,
                              hipStream_t stream) {
  (void)in_sizes; (void)n_in; (void)out_size; (void)ws_size;
  const float* enc  = (const float*)d_in[0];
  const float* Wp   = (const float*)d_in[1];
  const float* bp   = (const float*)d_in[2];
  const float* Wqk  = (const float*)d_in[3];   // (2,3,128,2048)
  const float* bqk  = (const float*)d_in[4];   // (2,2048)
  const float* Wv   = (const float*)d_in[5];   // (2,128,1024)
  const float* bv   = (const float*)d_in[6];
  const float* Wc   = (const float*)d_in[7];   // (2,1024,128)
  const float* bc   = (const float*)d_in[8];
  const float* W1   = (const float*)d_in[9];   // (2,128,512)
  const float* b1   = (const float*)d_in[10];
  const float* W2   = (const float*)d_in[11];  // (2,512,128)
  const float* b2   = (const float*)d_in[12];
  const float* g1   = (const float*)d_in[13];
  const float* be1  = (const float*)d_in[14];
  const float* g2   = (const float*)d_in[15];
  const float* be2  = (const float*)d_in[16];
  const float* Wout = (const float*)d_in[17];
  const float* bout = (const float*)d_in[18];

  float* out  = (float*)d_out;
  float* attn = out + ROWS_;                   // (B,2,H,1024,1024) f32

  // workspace carve-up (~88 MB total)
  char* ws = (char*)d_ws;
  float*    x    = (float*)ws;      ws += (size_t)ROWS_ * DM_ * 4;           // 4 MB
  _Float16* xh   = (_Float16*)ws;   ws += (size_t)ROWS_ * DM_ * 2;           // 2 MB
  _Float16* qb   = (_Float16*)ws;   ws += (size_t)B_ * NH_ * WIN_ * DM_ * 2; // 16 MB
  _Float16* kb   = (_Float16*)ws;   ws += (size_t)B_ * NH_ * WIN_ * DM_ * 2; // 16 MB
  _Float16* vt   = (_Float16*)ws;   ws += (size_t)B_ * NH_ * WIN_ * DM_ * 2; // 16 MB
  _Float16* ab   = (_Float16*)ws;   ws += (size_t)ROWS_ * DH_ * 2;           // 16 MB
  _Float16* hb   = (_Float16*)ws;   ws += (size_t)ROWS_ * DHID_ * 2;         //  8 MB
  float*    cbuf = (float*)ws;      ws += (size_t)ROWS_ * DM_ * 4;           //  4 MB
  _Float16* Wvt  = (_Float16*)ws;   ws += (size_t)NL_ * DH_ * DM_ * 2;       // 512 KB
  _Float16* Wqkt = (_Float16*)ws;   ws += (size_t)NL_ * 3 * 2 * DH_ * DM_ * 2; // 3 MB
  _Float16* Wct  = (_Float16*)ws;   ws += (size_t)NL_ * DM_ * DH_ * 2;       // 512 KB
  _Float16* W1t  = (_Float16*)ws;   ws += (size_t)NL_ * DHID_ * DM_ * 2;     // 256 KB
  _Float16* W2t  = (_Float16*)ws;   ws += (size_t)NL_ * DM_ * DHID_ * 2;     // 256 KB
  unsigned char* mask = (unsigned char*)ws;    /* 1 MB */

  const int TB = 256;                          // 8 waves per block

  mask_kernel <<<(WIN_ * WIN_) / TB, TB, 0, stream>>>(mask);
  embed_kernel<<<(ROWS_ * DM_) / TB, TB, 0, stream>>>(enc, Wp, bp, x, xh);

  // pre-convert + transpose all weights to f16 (N,K)
  for (int l = 0; l < NL_; ++l) {
    transpose_cvt_kernel<<<dim3(DH_/32, DM_/32), TB, 0, stream>>>(
        Wv + (size_t)l*DM_*DH_, Wvt + (size_t)l*DH_*DM_, DM_, DH_);
    for (int w = 0; w < 3; ++w)
      transpose_cvt_kernel<<<dim3(2*DH_/32, DM_/32), TB, 0, stream>>>(
          Wqk + (size_t)(l*3 + w)*DM_*2*DH_, Wqkt + (size_t)(l*3 + w)*2*DH_*DM_,
          DM_, 2*DH_);
    transpose_cvt_kernel<<<dim3(DM_/32, DH_/32), TB, 0, stream>>>(
        Wc + (size_t)l*DH_*DM_, Wct + (size_t)l*DM_*DH_, DH_, DM_);
    transpose_cvt_kernel<<<dim3(DHID_/32, DM_/32), TB, 0, stream>>>(
        W1 + (size_t)l*DM_*DHID_, W1t + (size_t)l*DHID_*DM_, DM_, DHID_);
    transpose_cvt_kernel<<<dim3(DM_/32, DHID_/32), TB, 0, stream>>>(
        W2 + (size_t)l*DHID_*DM_, W2t + (size_t)l*DM_*DHID_, DHID_, DM_);
  }

  for (int l = 0; l < NL_; ++l) {
    wmma_vproj <<<(ROWS_/16)*(DH_/16)/8,    TB, 0, stream>>>(
        xh, Wvt + (size_t)l*DH_*DM_, bv + (size_t)l*DH_, vt);
    wmma_qkproj<<<(ROWS_/16)*(2*DH_/16)/8,  TB, 0, stream>>>(
        xh, Wqkt + (size_t)l*3*2*DH_*DM_, bqk + (size_t)l*2*DH_, qb, kb);
    wmma_scores<<<(B_*NH_*4096)/8,          TB, 0, stream>>>(qb, kb, mask, attn, l);
    softmax_kernel<<<(B_*NH_*WIN_)/8,       TB, 0, stream>>>(attn, l);
    wmma_av    <<<(B_*NH_*512)/8,           TB, 0, stream>>>(attn, vt, ab, l);
    wmma_gemm_f32out<<<(ROWS_/16)*(DM_/16)/8, TB, 0, stream>>>(
        ab, Wct + (size_t)l*DM_*DH_, bc + (size_t)l*DM_, cbuf, ROWS_, DM_, DH_);
    add_ln_kernel<<<ROWS_/8, TB, 0, stream>>>(x, cbuf, g1 + l*DM_, be1 + l*DM_, xh);
    wmma_gemm_relu_f16out<<<(ROWS_/16)*(DHID_/16)/8, TB, 0, stream>>>(
        xh, W1t + (size_t)l*DHID_*DM_, b1 + (size_t)l*DHID_, hb, ROWS_, DHID_, DM_);
    wmma_gemm_f32out<<<(ROWS_/16)*(DM_/16)/8, TB, 0, stream>>>(
        hb, W2t + (size_t)l*DM_*DHID_, b2 + (size_t)l*DM_, cbuf, ROWS_, DM_, DHID_);
    add_ln_kernel<<<ROWS_/8, TB, 0, stream>>>(x, cbuf, g2 + l*DM_, be2 + l*DM_, xh);
  }

  out_kernel<<<ROWS_ / TB, TB, 0, stream>>>(x, Wout, bout, out);
}
